// RNN_8057358648148
// MI455X (gfx1250) — matrix operand order repaired and checked by reading, not compiled, not run on hardware
//
#include <hip/hip_runtime.h>

typedef __attribute__((ext_vector_type(16))) _Float16 v16h;
typedef __attribute__((ext_vector_type(8)))  float    v8f;
typedef __attribute__((ext_vector_type(4)))  float    v4f;
typedef __attribute__((ext_vector_type(2)))  float    v2f;

namespace {
constexpr int kSeq   = 2048;
constexpr int kBatch = 2048;
constexpr int kIn    = 16;
constexpr int kHid   = 64;
constexpr int kWaves = 4;   // waves per block
constexpr int kSRow  = 68;  // padded f32 staging row stride (dwords)
}

union V16H {
  v16h v;
  _Float16 h[16];
};

__device__ __forceinline__ float fast_tanh(float v) {
#if __has_builtin(__builtin_amdgcn_tanhf)
  return __builtin_amdgcn_tanhf(v);        // native v_tanh_f32 on gfx1250
#else
  float e = __builtin_amdgcn_exp2f(v * 2.8853900817779268f);
  return 1.0f - 2.0f * __builtin_amdgcn_rcpf(e + 1.0f);
#endif
}

// Transposed recurrence: G = h^T (64 x 16), one wave owns 16 batch columns.
//   G_new = tanh( [W | V | b] (64x96)  @  [G ; x^T ; 1-col layout] (96x16) )
// A (invariant, 96 VGPRs): M-tiles of [W | V | bias].  B (per step):
// kb=0,1 <- packed tanh outputs (lane-local!), kb=2 <- x fragment + ones.
// K map: 0..63 = hidden (K = hidden index in AMD table order, no column permutation
// so that WMMA D layout == next-step B layout per lane), 64..71 = x[0:8],
// 72..79 = x[8:16], 80 = ones (bias row in A), 81..87/88..95 = 0.
__global__ __launch_bounds__(kWaves * 32, 1)
void rnn_scan_wmma(const float* __restrict__ x,
                   const float* __restrict__ w_weight,
                   const float* __restrict__ w_bias,
                   const float* __restrict__ v_weight,
                   const float* __restrict__ v_bias,
                   float* __restrict__ out)
{
  __shared__ float sbuf_all[kWaves][16][kSRow];   // f32 h_seq staging, wave-private

  const int lane = threadIdx.x & 31;
  const int wib  = threadIdx.x >> 5;
  const int b0   = (blockIdx.x * kWaves + wib) * 16;   // batch base (B columns)
  float (*sbuf)[kSRow] = sbuf_all[wib];

  const int hsel = lane >> 4;   // 0: lanes 0-15, 1: lanes 16-31
  const int l15  = lane & 15;

  // ---- Invariant A fragments: 4 M-tiles x 3 K-blocks, kept in VGPRs ----
  // A-frag layout: halves j<8 -> K = kb*32+8*hsel+j ; j>=8 -> K = kb*32+8*hsel+16+(j-8)
  // So A holds W[m][K] directly, i.e. W[m][kb*32+8*hsel+j] (j<8), W[m][kb*32+16+8*hsel+(j-8)] (j>=8).
  V16H afr[3][4];
  #pragma unroll
  for (int mt = 0; mt < 4; ++mt) {
    const int m = mt * 16 + l15;                  // hidden-unit row
    #pragma unroll
    for (int kb = 0; kb < 2; ++kb) {
      const float* wr = w_weight + m * kHid + kb * 32 + hsel * 8;
      #pragma unroll
      for (int j = 0; j < 16; ++j) afr[kb][mt].h[j] = (_Float16)wr[(j < 8) ? j : j + 8];
    }
    // kb==2: lanes carry V[m][8*hsel+0..7] in j<8; hsel=0 lanes carry bias at j=8 (ones slot)
    const float* vr  = v_weight + m * kIn;
    const float bias = w_bias[m] + v_bias[m];
    #pragma unroll
    for (int j = 0; j < 16; ++j) {
      float wv  = (j < 8) ? vr[8 * hsel + j] : 0.0f;
      float alt = (j == 8 && !hsel) ? bias : 0.0f;
      afr[2][mt].h[j] = (_Float16)((j < 8) ? wv : alt);
    }
  }

  // ---- B kb=2 fragment: upper half loop-invariant (ones-column at K=80) ----
  V16H b2;
  b2.h[8] = hsel ? (_Float16)0.0f : (_Float16)1.0f;
  #pragma unroll
  for (int j = 9; j < 16; ++j) b2.h[j] = (_Float16)0.0f;

  // ---- hidden-state B operands, h0 = 0 ----
  V16H bh0 = {}, bh1 = {};

  // ---- software-pipelined x feed ----
  const float* xptr  = x + (size_t)(b0 + l15) * kIn + hsel * 8;
  const size_t xstep = (size_t)kBatch * kIn;
  v4f cx0 = *(const v4f*)(xptr);
  v4f cx1 = *(const v4f*)(xptr + 4);

  float*       orow  = out + (size_t)b0 * kHid;   // h_seq[0, b0, 0]
  const size_t ostep = (size_t)kBatch * kHid;

  #pragma unroll 1
  for (int s = 0; s < kSeq; ++s) {
    // ---- finish B kb=2 from current x ----
    #pragma unroll
    for (int j = 0; j < 4; ++j) b2.h[j]     = (_Float16)cx0[j];
    #pragma unroll
    for (int j = 0; j < 4; ++j) b2.h[4 + j] = (_Float16)cx1[j];

    // ---- issue next step's x loads (clamped), off the critical path ----
    if (s + 1 < kSeq) xptr += xstep;
    v4f nx0 = *(const v4f*)(xptr);
    v4f nx1 = *(const v4f*)(xptr + 4);
    __builtin_prefetch(xptr + xstep, 0, 1);       // speculative: safe if OOB

    // ---- 4 independent chains of 3 dependent WMMAs (A invariant!) ----
    v8f acc[4];
    #pragma unroll
    for (int t = 0; t < 4; ++t) {
      v8f c = {};
      c = __builtin_amdgcn_wmma_f32_16x16x32_f16(false, afr[0][t].v, false, bh0.v,
                                                 (short)0, c, false, false);
      c = __builtin_amdgcn_wmma_f32_16x16x32_f16(false, afr[1][t].v, false, bh1.v,
                                                 (short)0, c, false, false);
      c = __builtin_amdgcn_wmma_f32_16x16x32_f16(false, afr[2][t].v, false, b2.v,
                                                 (short)0, c, false, false);
      acc[t] = c;
    }

    // ---- tanh in place ----
    #pragma unroll
    for (int t = 0; t < 4; ++t)
      #pragma unroll
      for (int i = 0; i < 8; ++i) acc[t][i] = fast_tanh(acc[t][i]);

    // ---- lane-local repack into next step's B operands (no DS, no shuffles) ----
    #pragma unroll
    for (int e = 0; e < 2; ++e)
      #pragma unroll
      for (int i = 0; i < 8; ++i) {
        bh0.h[8 * e + i] = (_Float16)acc[e][i];       // kb=0 <- tiles 0,1
        bh1.h[8 * e + i] = (_Float16)acc[2 + e][i];   // kb=1 <- tiles 2,3
      }

    // ---- h_seq output: transpose via f32 LDS staging, stream b128 stores ----
    #pragma unroll
    for (int t = 0; t < 4; ++t)
      #pragma unroll
      for (int i = 0; i < 8; i += 2) {
        v2f p = { acc[t][i], acc[t][i + 1] };         // hidden m, m+1 (same lane)
        *(v2f*)&sbuf[l15][16 * t + 8 * hsel + i] = p; // sbuf[batch][hidden]
      }
    #pragma unroll
    for (int q = 0; q < 8; ++q) {
      const int r = 2 * q + hsel;                     // batch row within tile
      v4f d = *(const v4f*)&sbuf[r][l15 * 4];
      *(v4f*)&orow[(size_t)r * kHid + l15 * 4] = d;   // 512B contiguous / instr
    }
    orow += ostep;

    cx0 = nx0; cx1 = nx1;
    __builtin_amdgcn_wave_barrier();   // keep stage-stores ahead of next reuse
  }
}

extern "C" void kernel_launch(void* const* d_in, const int* in_sizes, int n_in,
                              void* d_out, int out_size, void* d_ws, size_t ws_size,
                              hipStream_t stream) {
  (void)in_sizes; (void)n_in; (void)d_ws; (void)ws_size; (void)out_size;
  const float* x        = (const float*)d_in[0];
  const float* w_weight = (const float*)d_in[1];
  const float* w_bias   = (const float*)d_in[2];
  const float* v_weight = (const float*)d_in[3];
  const float* v_bias   = (const float*)d_in[4];
  float* out = (float*)d_out;

  dim3 block(kWaves * 32);                // 128 threads = 4 waves (wave32)
  dim3 grid(kBatch / (16 * kWaves));      // 32 blocks -> 128 waves
  rnn_scan_wmma<<<grid, block, 0, stream>>>(x, w_weight, w_bias, v_weight, v_bias, out);

  // h (final hidden state) == h_seq[-1]: one exact D2D copy.
  const size_t slice = (size_t)kBatch * kHid * sizeof(float);
  float* last_slice  = out + (size_t)(kSeq - 1) * kBatch * kHid;
  float* hfin        = out + (size_t)kSeq * kBatch * kHid;
  hipMemcpyAsync(hfin, last_slice, slice, hipMemcpyDeviceToDevice, stream);
}
